// CompressedGPT2Attention_64587718198015
// MI455X (gfx1250) — matrix operand order repaired and checked by reading, not compiled, not run on hardware
//
#include <hip/hip_runtime.h>
#include <hip/hip_bf16.h>
#include <stdint.h>

// ---------------------------------------------------------------------------
// CompressedGPT2Attention on gfx1250 (MI455X), BF16 WMMA pipeline:
//   1) fp32 -> bf16 conversion of activations + weights
//   2) Q/K/V projection GEMMs via v_wmma_f32_16x16x32_bf16 (128x128 tile)
//      with async global->LDS staging of the A tile
//   3) causal flash attention: async-staged K tile, transposed V tile,
//      WMMA scores + PV, online softmax with wave32 butterflies
//   4) output GEMM (bf16 x bf16 -> fp32) into d_out
// ---------------------------------------------------------------------------

typedef __attribute__((ext_vector_type(16))) __bf16 v16bf;
typedef __attribute__((ext_vector_type(8)))  float  v8f;

union FragBF {
    v16bf v;
    uint4 q[2];
};

union Vec16B {
    uint4 v;
    unsigned short h[8];
};

__device__ __forceinline__ unsigned short f2bfu(float f) {
    unsigned u = __builtin_bit_cast(unsigned, f);
    unsigned r = u + 0x7FFFu + ((u >> 16) & 1u);   // round-to-nearest-even
    return (unsigned short)(r >> 16);
}

__device__ __forceinline__ v8f wmma_bf16(const FragBF& a, const FragBF& b, v8f c) {
    return __builtin_amdgcn_wmma_f32_16x16x32_bf16(
        /*neg_a=*/false, a.v, /*neg_b=*/false, b.v,
        /*c_mod=*/(short)0, c, /*reuse_a=*/false, /*reuse_b=*/false);
}

// Async global->LDS 16-byte copy (CDNA5 GLOBAL_LOAD_ASYNC_TO_LDS_B128, GV mode).
// ldsAddr: 32-bit LDS byte address (low 32 bits of flat shared pointer).
// gAddr  : 64-bit global byte address.
__device__ __forceinline__ void async_copy_b128(const void* ldsPtr, const void* gPtr) {
    unsigned lds = (unsigned)(uintptr_t)ldsPtr;
    unsigned long long ga = (unsigned long long)(uintptr_t)gPtr;
    asm volatile("global_load_async_to_lds_b128 %0, %1, off"
                 :: "v"(lds), "v"(ga) : "memory");
}

__device__ __forceinline__ void async_wait0() {
    asm volatile("s_wait_asynccnt 0" ::: "memory");
}

// ---------------------------------------------------------------------------
// fp32 -> bf16 (stored as u16) conversion
// ---------------------------------------------------------------------------
__global__ void cvt_f32_bf16(const float* __restrict__ src,
                             unsigned short* __restrict__ dst, int n) {
    int i = blockIdx.x * blockDim.x + threadIdx.x;
    int stride = gridDim.x * blockDim.x;
    for (; i < n; i += stride) dst[i] = f2bfu(src[i]);
}

// ---------------------------------------------------------------------------
// Tiled GEMM: C[M,N] = A[M,K](bf16) * B[K,N](bf16) + bias[N](f32)
// Block tile 128x128, BK=32, 256 threads (8 waves), wave tile 32x64
// (2x4 WMMA accumulators -> 8 WMMAs per K-step per wave).
// A tile staged via async global->LDS; B tile transposed manually.
// ---------------------------------------------------------------------------
template <bool OUT_F32>
__global__ __launch_bounds__(256) void gemm_bf16_wmma(
    const unsigned short* __restrict__ A,
    const unsigned short* __restrict__ Bw,
    const float* __restrict__ bias,
    void* __restrict__ Cout,
    int M, int N, int K) {

    constexpr int BM = 128, BN = 128, BK = 32, PAD = 8;
    __shared__ unsigned short As[BM][BK + PAD];   // row-major tile
    __shared__ unsigned short Bs[BN][BK + PAD];   // transposed: [n][k]

    const int tid  = threadIdx.x;
    const int wave = tid >> 5;
    const int lane = tid & 31;
    const int lm   = lane & 15;
    const int half = lane >> 4;
    const int wr   = wave & 3;   // M sub-tile (32 rows)
    const int wc   = wave >> 2;  // N sub-tile (64 cols)
    const int bm0  = blockIdx.y * BM;
    const int bn0  = blockIdx.x * BN;

    v8f acc[2][4] = {};

    for (int k0 = 0; k0 < K; k0 += BK) {
        // --- stage A tile: 128x32 bf16, 2 async b128 copies per thread ---
        #pragma unroll
        for (int it = 0; it < 2; ++it) {
            int idx = (tid + it * 256) * 8;
            int r = idx >> 5, c = idx & 31;
            const unsigned short* src = A + (size_t)(bm0 + r) * K + k0 + c;
            async_copy_b128(&As[r][c], src);
            if (k0 + BK < K)  // prefetch next K tile (global_prefetch_b8)
                __builtin_prefetch(src + BK, 0, 0);
        }
        // --- stage B tile transposed: read [k][n] rows, write [n][k] -----
        #pragma unroll
        for (int it = 0; it < 2; ++it) {
            int idx = (tid + it * 256) * 8;
            int kr = idx >> 7, nc = idx & 127;
            Vec16B d;
            d.v = *(const uint4*)(Bw + (size_t)(k0 + kr) * N + bn0 + nc);
            #pragma unroll
            for (int j = 0; j < 8; ++j) Bs[nc + j][kr] = d.h[j];
        }
        async_wait0();
        __syncthreads();

        // --- build fragments (2 x b128 each) & WMMA ----------------------
        FragBF a[2], b[4];
        #pragma unroll
        for (int i = 0; i < 2; ++i) {
            const unsigned short* row = &As[wr * 32 + i * 16 + lm][0];
            a[i].q[0] = *(const uint4*)(row + half * 8);
            a[i].q[1] = *(const uint4*)(row + 16 + half * 8);
        }
        #pragma unroll
        for (int j = 0; j < 4; ++j) {
            const unsigned short* row = &Bs[wc * 64 + j * 16 + lm][0];
            b[j].q[0] = *(const uint4*)(row + half * 8);
            b[j].q[1] = *(const uint4*)(row + 16 + half * 8);
        }
        #pragma unroll
        for (int i = 0; i < 2; ++i)
            #pragma unroll
            for (int j = 0; j < 4; ++j)
                acc[i][j] = wmma_bf16(a[i], b[j], acc[i][j]);
        __syncthreads();
    }

    // --- store (C layout: lane%16 = col, VGPR r = row, +8 for upper half) --
    #pragma unroll
    for (int i = 0; i < 2; ++i) {
        #pragma unroll
        for (int j = 0; j < 4; ++j) {
            int gc = bn0 + wc * 64 + j * 16 + lm;
            float bv = bias[gc];
            #pragma unroll
            for (int r = 0; r < 8; ++r) {
                int gr = bm0 + wr * 32 + i * 16 + half * 8 + r;
                float val = acc[i][j][r] + bv;
                if (OUT_F32)
                    ((float*)Cout)[(size_t)gr * N + gc] = val;
                else
                    ((unsigned short*)Cout)[(size_t)gr * N + gc] = f2bfu(val);
            }
        }
    }
}

// ---------------------------------------------------------------------------
// Causal flash attention.
//   Q,K: [B,S,E] bf16 (head h at column h*HD), V: [B,S,H*R] bf16.
//   Out: attn [B,S,H*R] bf16.
// Block = 128 threads (4 waves); grid = (S/64, B*H).
// K tile (32x64, async copy) and transposed V tile (32x32) staged in LDS
// once per key block; each wave owns 16 query rows.
// ---------------------------------------------------------------------------
__global__ __launch_bounds__(128) void attn_wmma(
    const unsigned short* __restrict__ Q,
    const unsigned short* __restrict__ Km,
    const unsigned short* __restrict__ V,
    unsigned short* __restrict__ Aout) {

    constexpr int S = 2048, E = 1024, H = 16, HD = 64, R = 32;
    constexpr int KPAD = 8, VPAD = 8, PPAD = 8;
    __shared__ unsigned short Ks[32][HD + KPAD];       // [key][d]
    __shared__ unsigned short Vs[R][32 + VPAD];        // transposed [col][key]
    __shared__ unsigned short Pbuf[4][16][R + PPAD];   // per-wave P tile

    const int tid  = threadIdx.x;
    const int w    = tid >> 5;
    const int lane = tid & 31;
    const int lm   = lane & 15;
    const int half = lane >> 4;
    const int bh   = blockIdx.y;
    const int b    = bh / H;
    const int h    = bh % H;
    const int q0b  = blockIdx.x * 64;
    const int q0   = q0b + w * 16;

    // --- load Q as two A-fragments (d = 0..31, 32..63), kept in registers --
    FragBF qa[2];
    {
        const unsigned short* qrow = Q + ((size_t)(b * S + q0 + lm)) * E + h * HD;
        #pragma unroll
        for (int kb = 0; kb < 2; ++kb) {
            qa[kb].q[0] = *(const uint4*)(qrow + kb * 32 + half * 8);
            qa[kb].q[1] = *(const uint4*)(qrow + kb * 32 + 16 + half * 8);
        }
    }

    float m_i[8], l_i[8];
    #pragma unroll
    for (int r = 0; r < 8; ++r) { m_i[r] = -3.0e38f; l_i[r] = 0.0f; }
    v8f o[2] = {};

    const float scale = 0.125f;  // 1/sqrt(64)
    const int jmax_w = (q0 + 15) >> 5;          // this wave's causal limit
    const int jmax_b = (q0b + 63) >> 5;         // block-uniform loop bound

    for (int j = 0; j <= jmax_b; ++j) {
        const int kbase = j * 32;

        // ---- cooperative staging: K tile (async), V tile transposed -----
        {
            int key = tid >> 2;                  // 4 threads per key row
            int d0  = (tid & 3) * 16;            // 16 dims each
            const unsigned short* ksrc =
                Km + ((size_t)(b * S + kbase + key)) * E + h * HD + d0;
            async_copy_b128(&Ks[key][d0],     ksrc);
            async_copy_b128(&Ks[key][d0 + 8], ksrc + 8);

            int c0 = (tid & 3) * 8;              // 8 cols each
            Vec16B vd;
            vd.v = *(const uint4*)(V + ((size_t)(b * S + kbase + key)) * (H * R)
                                     + h * R + c0);
            #pragma unroll
            for (int i = 0; i < 8; ++i) Vs[c0 + i][key] = vd.h[i];
        }
        async_wait0();
        __syncthreads();

        if (j <= jmax_w) {
            // ---- scores: two 16-key tiles, each 2 WMMA K-steps over HD=64 --
            float sc[2][8];
            #pragma unroll
            for (int t = 0; t < 2; ++t) {
                FragBF kbf[2];
                const unsigned short* krow = &Ks[t * 16 + lm][0];
                #pragma unroll
                for (int kb = 0; kb < 2; ++kb) {
                    kbf[kb].q[0] = *(const uint4*)(krow + kb * 32 + half * 8);
                    kbf[kb].q[1] = *(const uint4*)(krow + kb * 32 + 16 + half * 8);
                }
                v8f s = {};
                s = wmma_bf16(qa[0], kbf[0], s);
                s = wmma_bf16(qa[1], kbf[1], s);
                #pragma unroll
                for (int r = 0; r < 8; ++r) {
                    int qrow = q0 + half * 8 + r;
                    int kcol = kbase + t * 16 + lm;
                    float sv = s[r] * scale;
                    sc[t][r] = (kcol > qrow) ? -3.0e38f : sv;
                }
            }

            // ---- online softmax (butterfly within 16-lane halves) --------
            #pragma unroll
            for (int r = 0; r < 8; ++r) {
                float rmax = fmaxf(sc[0][r], sc[1][r]);
                #pragma unroll
                for (int m = 1; m <= 8; m <<= 1)
                    rmax = fmaxf(rmax, __shfl_xor(rmax, m, 32));
                float mnew  = fmaxf(m_i[r], rmax);
                float alpha = __expf(m_i[r] - mnew);
                float p0 = __expf(sc[0][r] - mnew);
                float p1 = __expf(sc[1][r] - mnew);
                Pbuf[w][half * 8 + r][lm]      = f2bfu(p0);
                Pbuf[w][half * 8 + r][16 + lm] = f2bfu(p1);
                float rsum = p0 + p1;
                #pragma unroll
                for (int m = 1; m <= 8; m <<= 1)
                    rsum += __shfl_xor(rsum, m, 32);
                l_i[r] = l_i[r] * alpha + rsum;
                m_i[r] = mnew;
                o[0][r] *= alpha;
                o[1][r] *= alpha;
            }

            // ---- P (A-frag from LDS) x V (B-frag from LDS) ---------------
            FragBF pa;
            {
                const unsigned short* prow = &Pbuf[w][lm][0];
                pa.q[0] = *(const uint4*)(prow + half * 8);
                pa.q[1] = *(const uint4*)(prow + 16 + half * 8);
            }
            #pragma unroll
            for (int n16 = 0; n16 < 2; ++n16) {
                FragBF vb;
                const unsigned short* vrow = &Vs[n16 * 16 + lm][0];
                vb.q[0] = *(const uint4*)(vrow + half * 8);
                vb.q[1] = *(const uint4*)(vrow + 16 + half * 8);
                o[n16] = wmma_bf16(pa, vb, o[n16]);
            }
        }
        __syncthreads();
    }

    // ---- normalize and store attn [B,S,H*R] ------------------------------
    #pragma unroll
    for (int n16 = 0; n16 < 2; ++n16)
        #pragma unroll
        for (int r = 0; r < 8; ++r) {
            int row = q0 + half * 8 + r;
            int col = n16 * 16 + lm;
            float val = o[n16][r] / l_i[r];
            Aout[((size_t)(b * S + row)) * (H * R) + h * R + col] = f2bfu(val);
        }
}

// ---------------------------------------------------------------------------
// Host launch
// ---------------------------------------------------------------------------
extern "C" void kernel_launch(void* const* d_in, const int* in_sizes, int n_in,
                              void* d_out, int out_size, void* d_ws, size_t ws_size,
                              hipStream_t stream) {
    constexpr int B = 4, S = 2048, E = 1024, H = 16;
    constexpr int M = B * S;          // 8192
    constexpr int HR = 512;           // H * R

    const float* X  = (const float*)d_in[0];
    const float* Wq = (const float*)d_in[1];
    const float* bq = (const float*)d_in[2];
    const float* Wk = (const float*)d_in[3];
    const float* bk = (const float*)d_in[4];
    const float* Wv = (const float*)d_in[5];
    const float* bv = (const float*)d_in[6];
    const float* Wo = (const float*)d_in[7];
    const float* bo = (const float*)d_in[8];

    unsigned short* ws = (unsigned short*)d_ws;
    size_t off = 0;
    unsigned short* Xbf = ws + off; off += (size_t)M * E;
    unsigned short* Wqb = ws + off; off += (size_t)E * E;
    unsigned short* Wkb = ws + off; off += (size_t)E * E;
    unsigned short* Wvb = ws + off; off += (size_t)E * HR;
    unsigned short* Wob = ws + off; off += (size_t)HR * E;
    unsigned short* Qbf = ws + off; off += (size_t)M * E;
    unsigned short* Kbf = ws + off; off += (size_t)M * E;
    unsigned short* Vbf = ws + off; off += (size_t)M * HR;
    unsigned short* Abf = ws + off; off += (size_t)M * HR;

    // 1) conversions
    cvt_f32_bf16<<<1024, 256, 0, stream>>>(X,  Xbf, M * E);
    cvt_f32_bf16<<<512,  256, 0, stream>>>(Wq, Wqb, E * E);
    cvt_f32_bf16<<<512,  256, 0, stream>>>(Wk, Wkb, E * E);
    cvt_f32_bf16<<<256,  256, 0, stream>>>(Wv, Wvb, E * HR);
    cvt_f32_bf16<<<256,  256, 0, stream>>>(Wo, Wob, HR * E);

    // 2) projections (bf16 out), 128x128 block tiles
    gemm_bf16_wmma<false><<<dim3(E / 128,  M / 128), 256, 0, stream>>>(Xbf, Wqb, bq, Qbf, M, E,  E);
    gemm_bf16_wmma<false><<<dim3(E / 128,  M / 128), 256, 0, stream>>>(Xbf, Wkb, bk, Kbf, M, E,  E);
    gemm_bf16_wmma<false><<<dim3(HR / 128, M / 128), 256, 0, stream>>>(Xbf, Wvb, bv, Vbf, M, HR, E);

    // 3) causal flash attention
    attn_wmma<<<dim3(S / 64, B * H), 128, 0, stream>>>(Qbf, Kbf, Vbf, Abf);

    // 4) output GEMM (fp32 out)
    gemm_bf16_wmma<true><<<dim3(E / 128, M / 128), 256, 0, stream>>>(Abf, Wob, bo, d_out, M, E, HR);
}
